// CausalSelfAttention_257698038387
// MI455X (gfx1250) — matrix operand order repaired and checked by reading, not compiled, not run on hardware
//
#include <hip/hip_runtime.h>

#define HIDDEN 1024
#define HEADS  16
#define HDIM   64
#define BATCH  2
#define SEQ    2048
#define M_TOT  (BATCH * SEQ)   // 4096 rows

typedef __bf16 bf16;
typedef bf16  v16bf __attribute__((ext_vector_type(16)));
typedef bf16  v8bf  __attribute__((ext_vector_type(8)));
typedef float v8f   __attribute__((ext_vector_type(8)));
typedef unsigned int v2u __attribute__((ext_vector_type(2)));
typedef float v4f  __attribute__((ext_vector_type(4)));

// ---------- helpers ----------

static __device__ __forceinline__ unsigned f2bf_round(float f) {
    unsigned u = __builtin_bit_cast(unsigned, f);
    return (u + 0x7FFFu + ((u >> 16) & 1u)) >> 16;   // RNE
}
static __device__ __forceinline__ unsigned short f2bf_bits(float f) {
    return (unsigned short)f2bf_round(f);
}

// A-fragment 16x32 bf16 (ISA 7.12.2): lane L -> row m0+(L&15);
// lanes 0-15 hold K {k0..k0+7, k0+16..k0+23}; lanes 16-31 hold K {+8..+15, +24..+31}
static __device__ __forceinline__ v16bf load_a_frag(const unsigned short* base, int ld,
                                                    int m0, int k0, int lane) {
    int row = m0 + (lane & 15);
    int kb  = k0 + ((lane >> 4) << 3);
    const unsigned short* p = base + (size_t)row * ld + kb;
    v8bf lo = *(const v8bf*)(p);
    v8bf hi = *(const v8bf*)(p + 16);
    v16bf a;
#pragma unroll
    for (int i = 0; i < 8; ++i) { a[i] = lo[i]; a[8 + i] = hi[i]; }
    return a;
}

// B-fragment 32x16 bf16: lane L -> column n0+(L&15);
// lanes 0-15 hold K {k0..k0+7, k0+16..k0+23}; lanes 16-31 hold K {+8..+15, +24..+31}.
// base is row-major [n][k] (i.e. B[k][n] = base[n][k]) -- matches x @ W.T directly.
static __device__ __forceinline__ v16bf load_b_frag(const unsigned short* base, int ld,
                                                    int n0, int k0, int lane) {
    int n  = n0 + (lane & 15);
    int kb = k0 + ((lane >> 4) << 3);
    const unsigned short* p = base + (size_t)n * ld + kb;
    v8bf lo = *(const v8bf*)(p);
    v8bf hi = *(const v8bf*)(p + 16);
    v16bf b;
#pragma unroll
    for (int i = 0; i < 8; ++i) { b[i] = lo[i]; b[8 + i] = hi[i]; }
    return b;
}

static __device__ __forceinline__ const void* b_frag_addr(const unsigned short* base,
                                                          int ld, int n0, int k0, int lane) {
    int n  = n0 + (lane & 15);
    int kb = k0 + ((lane >> 4) << 4);
    return (const void*)(base + (size_t)n * ld + kb);
}

static __device__ __forceinline__ v8f wmma_bf16(v16bf a, v16bf b, v8f c) {
    return __builtin_amdgcn_wmma_f32_16x16x32_bf16(false, a, false, b, (short)0, c,
                                                   false, false);
}

// ---------- kernels ----------

// fp32 -> bf16, 4 elements per thread, packed 64-bit stores
__global__ __launch_bounds__(256) void cvt_f32_bf16(const float* __restrict__ in,
                                                    unsigned int* __restrict__ out,
                                                    int n4) {
    int i = blockIdx.x * blockDim.x + threadIdx.x;
    if (i < n4) {
        v4f v = *(const v4f*)(in + (size_t)i * 4);
        v2u p;
        p[0] = f2bf_round(v[0]) | (f2bf_round(v[1]) << 16);
        p[1] = f2bf_round(v[2]) | (f2bf_round(v[3]) << 16);
        *(v2u*)(out + (size_t)i * 2) = p;
    }
}

// Fused QKV projection: one wave per 32x64 output tile (B-fragments reused across
// two M-subtiles -> 24 WMMAs per K-step for 12 B loads + 2 A loads).
// Writes Q,K as [B,H,T,D] bf16 and V transposed [B,H,D,T].
__global__ __launch_bounds__(32) void qkv_gemm(const unsigned short* __restrict__ xb,
                                               const unsigned short* __restrict__ wq,
                                               const unsigned short* __restrict__ wk,
                                               const unsigned short* __restrict__ wv,
                                               unsigned short* __restrict__ qo,
                                               unsigned short* __restrict__ ko,
                                               unsigned short* __restrict__ vt) {
    int lane = threadIdx.x;
    int ntn  = HIDDEN / 64;                 // 16 N-tiles
    int m0   = (blockIdx.x / ntn) * 32;
    int n0   = (blockIdx.x % ntn) * 64;

    v8f cq[2][4] = {}, ck[2][4] = {}, cv[2][4] = {};
    for (int kk = 0; kk < HIDDEN; kk += 32) {
        v16bf a0 = load_a_frag(xb, HIDDEN, m0,      kk, lane);
        v16bf a1 = load_a_frag(xb, HIDDEN, m0 + 16, kk, lane);
        if (kk + 32 < HIDDEN) {             // prefetch next weight rows
            __builtin_prefetch(b_frag_addr(wq, HIDDEN, n0, kk + 32, lane), 0, 1);
            __builtin_prefetch(b_frag_addr(wk, HIDDEN, n0, kk + 32, lane), 0, 1);
            __builtin_prefetch(b_frag_addr(wv, HIDDEN, n0, kk + 32, lane), 0, 1);
        }
#pragma unroll
        for (int s = 0; s < 4; ++s) {
            v16bf bq = load_b_frag(wq, HIDDEN, n0 + 16 * s, kk, lane);
            cq[0][s] = wmma_bf16(a0, bq, cq[0][s]);
            cq[1][s] = wmma_bf16(a1, bq, cq[1][s]);
            v16bf bk = load_b_frag(wk, HIDDEN, n0 + 16 * s, kk, lane);
            ck[0][s] = wmma_bf16(a0, bk, ck[0][s]);
            ck[1][s] = wmma_bf16(a1, bk, ck[1][s]);
            v16bf bv = load_b_frag(wv, HIDDEN, n0 + 16 * s, kk, lane);
            cv[0][s] = wmma_bf16(a0, bv, cv[0][s]);
            cv[1][s] = wmma_bf16(a1, bv, cv[1][s]);
        }
    }
    int h = n0 / HDIM;                      // 64-wide tile => single head
    int half = lane >> 4, l = lane & 15;
#pragma unroll
    for (int mt = 0; mt < 2; ++mt) {
#pragma unroll
        for (int s = 0; s < 4; ++s) {
#pragma unroll
            for (int r = 0; r < 8; ++r) {
                int m = m0 + 16 * mt + r + half * 8;
                int b = m / SEQ, t = m % SEQ;
                int d = 16 * s + l;
                size_t qi = ((size_t)(b * HEADS + h) * SEQ + t) * HDIM + d;
                qo[qi] = f2bf_bits(cq[mt][s][r]);
                ko[qi] = f2bf_bits(ck[mt][s][r]);
                size_t vi = ((size_t)(b * HEADS + h) * HDIM + d) * SEQ + t;
                vt[vi] = f2bf_bits(cv[mt][s][r]);
            }
        }
    }
}

// Flash attention: one wave per (b,h,16-query-row tile). Q A-fragments hoisted
// into registers (loop-invariant). Key tiles of 32: 4 score WMMAs, LDS-staged
// online softmax, 4 PV WMMAs; prefetch next K/V tiles.
__global__ __launch_bounds__(32) void flash_attn(const unsigned short* __restrict__ q,
                                                 const unsigned short* __restrict__ kmat,
                                                 const unsigned short* __restrict__ vt,
                                                 unsigned short* __restrict__ aout) {
    __shared__ float          lds_s[16][32];
    __shared__ unsigned short lds_p[16][32];
    __shared__ float          lds_alpha[16];
    __shared__ float          lds_inv[16];

    int lane = threadIdx.x;
    int half = lane >> 4, l = lane & 15;
    int blk = blockIdx.x;
    int qt  = blk & (SEQ / 16 - 1);         // 128 query tiles
    int bh  = blk >> 7;
    int q0  = qt * 16;

    const unsigned short* Q  = q    + (size_t)bh * SEQ * HDIM;
    const unsigned short* K  = kmat + (size_t)bh * SEQ * HDIM;
    const unsigned short* Vt = vt   + (size_t)bh * HDIM * SEQ;

    // loop-invariant Q fragments (16x64 tile = 2 A-fragments)
    v16bf qa0 = load_a_frag(Q, HDIM, q0, 0,  lane);
    v16bf qa1 = load_a_frag(Q, HDIM, q0, 32, lane);

    v8f o[4] = {};
    float rm = -1e30f, rs = 0.0f;           // row stats, valid in lanes 0-15
    const float scale = 0.125f;             // 1/sqrt(64)

    int nkt = (q0 + 16 + 31) >> 5;          // key tiles covering cols 0..q0+15
    for (int j = 0; j < nkt; ++j) {
        int jc = j * 32;
        if (j + 1 < nkt) {                  // prefetch next K/V tiles
            __builtin_prefetch(b_frag_addr(K, HDIM, jc + 32, 0, lane), 0, 1);
            __builtin_prefetch(b_frag_addr(Vt, SEQ, 0, jc + 32, lane), 0, 1);
            __builtin_prefetch(b_frag_addr(Vt, SEQ, 32, jc + 32, lane), 0, 1);
        }
        v8f s0 = {}, s1 = {};
        {
            v16bf b0 = load_b_frag(K, HDIM, jc, 0, lane);
            s0 = wmma_bf16(qa0, b0, s0);
            v16bf b1 = load_b_frag(K, HDIM, jc, 32, lane);
            s0 = wmma_bf16(qa1, b1, s0);
            v16bf b2 = load_b_frag(K, HDIM, jc + 16, 0, lane);
            s1 = wmma_bf16(qa0, b2, s1);
            v16bf b3 = load_b_frag(K, HDIM, jc + 16, 32, lane);
            s1 = wmma_bf16(qa1, b3, s1);
        }
        // scale + causal mask + stage scores in LDS
#pragma unroll
        for (int r = 0; r < 8; ++r) {
            int row = r + half * 8;
            int qpos = q0 + row;
            lds_s[row][l]      = (jc + l      <= qpos) ? s0[r] * scale : -1e30f;
            lds_s[row][16 + l] = (jc + 16 + l <= qpos) ? s1[r] * scale : -1e30f;
        }
        __syncthreads();
        if (lane < 16) {                    // one row per lane
            int row = lane;
            float mloc = -1e30f;
#pragma unroll
            for (int c = 0; c < 32; ++c) mloc = fmaxf(mloc, lds_s[row][c]);
            float mnew  = fmaxf(rm, mloc);
            float alpha = __expf(rm - mnew);
            float psum  = 0.0f;
#pragma unroll
            for (int c = 0; c < 32; ++c) {
                float p = __expf(lds_s[row][c] - mnew);
                psum += p;
                lds_p[row][c] = f2bf_bits(p);
            }
            rs = rs * alpha + psum;
            rm = mnew;
            lds_alpha[row] = alpha;
        }
        __syncthreads();
        // rescale accumulators and accumulate P @ V  (P reloaded as A-fragment)
        v16bf pa = load_a_frag(&lds_p[0][0], 32, 0, 0, lane);
#pragma unroll
        for (int s = 0; s < 4; ++s) {
#pragma unroll
            for (int r = 0; r < 8; ++r) o[s][r] *= lds_alpha[r + half * 8];
            v16bf bv = load_b_frag(Vt, SEQ, 16 * s, jc, lane);
            o[s] = wmma_bf16(pa, bv, o[s]);
        }
        __syncthreads();
    }
    if (lane < 16) lds_inv[lane] = 1.0f / rs;
    __syncthreads();

    int b = bh / HEADS, h = bh % HEADS;
#pragma unroll
    for (int s = 0; s < 4; ++s) {
#pragma unroll
        for (int r = 0; r < 8; ++r) {
            int row = r + half * 8;
            float val = o[s][r] * lds_inv[row];
            int t = q0 + row;
            size_t idx = ((size_t)(b * SEQ + t)) * HIDDEN + h * HDIM + 16 * s + l;
            aout[idx] = f2bf_bits(val);
        }
    }
}

// Output projection: y = attn_out @ Wo.T -> fp32; 32x64 tile per wave.
__global__ __launch_bounds__(32) void out_gemm(const unsigned short* __restrict__ ain,
                                               const unsigned short* __restrict__ wo,
                                               float* __restrict__ out) {
    int lane = threadIdx.x;
    int ntn  = HIDDEN / 64;
    int m0   = (blockIdx.x / ntn) * 32;
    int n0   = (blockIdx.x % ntn) * 64;
    v8f c[2][4] = {};
    for (int kk = 0; kk < HIDDEN; kk += 32) {
        v16bf a0 = load_a_frag(ain, HIDDEN, m0,      kk, lane);
        v16bf a1 = load_a_frag(ain, HIDDEN, m0 + 16, kk, lane);
        if (kk + 32 < HIDDEN)
            __builtin_prefetch(b_frag_addr(wo, HIDDEN, n0, kk + 32, lane), 0, 1);
#pragma unroll
        for (int s = 0; s < 4; ++s) {
            v16bf b = load_b_frag(wo, HIDDEN, n0 + 16 * s, kk, lane);
            c[0][s] = wmma_bf16(a0, b, c[0][s]);
            c[1][s] = wmma_bf16(a1, b, c[1][s]);
        }
    }
    int half = lane >> 4, l = lane & 15;
#pragma unroll
    for (int mt = 0; mt < 2; ++mt)
#pragma unroll
        for (int s = 0; s < 4; ++s)
#pragma unroll
            for (int r = 0; r < 8; ++r)
                out[(size_t)(m0 + 16 * mt + r + half * 8) * HIDDEN + n0 + 16 * s + l] =
                    c[mt][s][r];
}

// ---------- launch ----------

extern "C" void kernel_launch(void* const* d_in, const int* in_sizes, int n_in,
                              void* d_out, int out_size, void* d_ws, size_t ws_size,
                              hipStream_t stream) {
    const float* x  = (const float*)d_in[0];
    const float* Wq = (const float*)d_in[1];
    const float* Wk = (const float*)d_in[2];
    const float* Wv = (const float*)d_in[3];
    const float* Wo = (const float*)d_in[4];
    float* out = (float*)d_out;

    const size_t NX = (size_t)M_TOT * HIDDEN;       // 4 Mi elems
    const size_t NW = (size_t)HIDDEN * HIDDEN;      // 1 Mi elems
    unsigned short* ws = (unsigned short*)d_ws;
    unsigned short* xb  = ws;                        // 8 MiB
    unsigned short* wqb = xb  + NX;                  // 2 MiB each
    unsigned short* wkb = wqb + NW;
    unsigned short* wvb = wkb + NW;
    unsigned short* wob = wvb + NW;
    unsigned short* qb  = wob + NW;                  // 8 MiB each
    unsigned short* kb  = qb  + NX;
    unsigned short* vtb = kb  + NX;
    unsigned short* ab  = vtb + NX;                  // attn out, 8 MiB

    cvt_f32_bf16<<<(int)((NX / 4 + 255) / 256), 256, 0, stream>>>(x,  (unsigned int*)xb,  (int)(NX / 4));
    cvt_f32_bf16<<<(int)((NW / 4 + 255) / 256), 256, 0, stream>>>(Wq, (unsigned int*)wqb, (int)(NW / 4));
    cvt_f32_bf16<<<(int)((NW / 4 + 255) / 256), 256, 0, stream>>>(Wk, (unsigned int*)wkb, (int)(NW / 4));
    cvt_f32_bf16<<<(int)((NW / 4 + 255) / 256), 256, 0, stream>>>(Wv, (unsigned int*)wvb, (int)(NW / 4));
    cvt_f32_bf16<<<(int)((NW / 4 + 255) / 256), 256, 0, stream>>>(Wo, (unsigned int*)wob, (int)(NW / 4));

    int gemm_blocks = (M_TOT / 32) * (HIDDEN / 64); // 2048
    qkv_gemm<<<gemm_blocks, 32, 0, stream>>>(xb, wqb, wkb, wvb, qb, kb, vtb);

    int attn_blocks = BATCH * HEADS * (SEQ / 16);   // 4096
    flash_attn<<<attn_blocks, 32, 0, stream>>>(qb, kb, vtb, ab);

    out_gemm<<<gemm_blocks, 32, 0, stream>>>(ab, wob, out);
}
